// ThisGPT2Block_6330781794813
// MI455X (gfx1250) — matrix operand order repaired and checked by reading, not compiled, not run on hardware
//
#include <hip/hip_runtime.h>

// ---------------------------------------------------------------------------
// GPT-2 block on gfx1250: all GEMMs + attention through v_wmma_f32_16x16x32_bf16,
// GEMM tiles staged into LDS by the Tensor Data Mover (tensor_load_to_lds,
// TENSORcnt double-buffered). Weights pre-transposed so A and B tiles share one
// TDM descriptor shape (128 rows x 32 bf16, row pitch 64B data + 16B pad).
// ---------------------------------------------------------------------------

typedef __attribute__((ext_vector_type(16))) __bf16 v16bf;
typedef __attribute__((ext_vector_type(8)))  float  v8f;
typedef __attribute__((ext_vector_type(4)))  unsigned int v4u;
typedef __attribute__((ext_vector_type(8)))  int    v8i;
typedef __attribute__((ext_vector_type(4)))  int    v4i;

typedef unsigned short u16;
typedef unsigned int   u32;

#if __has_builtin(__builtin_amdgcn_tensor_load_to_lds)
#define HAVE_TDM 1
#if __has_include(<hip/amd_detail/amd_gfx1250_TDM.h>)
#define TDM_SIX_ARGS 1   // therock-10.0 headers -> 6-arg builtin
#else
#define TDM_SIX_ARGS 0   // ROCm 7.2 clang-22 -> 5-arg builtin
#endif
#else
#define HAVE_TDM 0
#endif

union Frag {
  v16bf v;
  u16   h[16];
  uint4 q[2];
};

__device__ __forceinline__ u16 f2bf(float f) {
  union { float f; u32 u; } c; c.f = f;
  u32 u = c.u + 0x7FFFu + ((c.u >> 16) & 1u);  // round-to-nearest-even
  return (u16)(u >> 16);
}

__device__ __forceinline__ v8f wmma_bf16(const Frag& a, const Frag& b, v8f c) {
  return __builtin_amdgcn_wmma_f32_16x16x32_bf16(false, a.v, false, b.v,
                                                 (short)0, c, false, false);
}

#if HAVE_TDM
// Issue a TDM load of a 128x32 bf16 tile (row stride = krow elements) into LDS
// at byte offset lds_addr, with LDS row pitch 64B data + 16B pad (-> [128][40]).
__device__ __forceinline__ void tdm_load_tile_128x32(u32 lds_addr, const u16* gp,
                                                     int krow, int rows) {
  const unsigned long long ga = (unsigned long long)(uintptr_t)gp;
  v4u g0;
  g0[0] = 1u;                                             // count=1 (user D#)
  g0[1] = lds_addr;                                       // lds_addr[31:0]
  g0[2] = (u32)ga;                                        // global_addr[31:0]
  g0[3] = (u32)((ga >> 32) & 0x01FFFFFFu) | (2u << 30);   // addr[56:32] | type=2
  v8i g1;
  // data_size=1(2B) | pad_enable | pad_interval=3(16 dw=64B) | pad_amount=3(4 dw=16B)
  g1[0] = (int)((1u << 16) | (1u << 20) | (3u << 22) | (3u << 25));
  g1[1] = (int)(((u32)krow & 0xFFFFu) << 16);             // tensor_dim0 lo16
  g1[2] = (int)(((u32)krow >> 16) | (((u32)rows & 0xFFFFu) << 16)); // dim0 hi | dim1 lo
  g1[3] = (int)(((u32)rows >> 16) | (32u << 16));         // dim1 hi | tile_dim0=32
  g1[4] = 128;                                            // tile_dim1=128, tile_dim2=0
  g1[5] = krow;                                           // tensor_dim0_stride lo32
  g1[6] = 0;                                              // stride hi | dim1_stride lo
  g1[7] = 0;
  const v4i gz = {0, 0, 0, 0};
#if TDM_SIX_ARGS
  const v8i gz8 = {0, 0, 0, 0, 0, 0, 0, 0};
  __builtin_amdgcn_tensor_load_to_lds(g0, g1, gz, gz, gz8, 0);
#else
  __builtin_amdgcn_tensor_load_to_lds(g0, g1, gz, gz, 0);
#endif
}
#endif

// ---------------------------------------------------------------------------
// Fused fp32->bf16 cast + transpose for weights: W[K][N] -> Wt[N][K]
// ---------------------------------------------------------------------------
__global__ __launch_bounds__(256) void transpose_cast_kernel(
    const float* __restrict__ W, u16* __restrict__ Wt, int K, int N) {
  __shared__ u16 T[32][33];
  const int tx = threadIdx.x & 31, ty = threadIdx.x >> 5;  // ty 0..7
  const int k0 = blockIdx.y * 32, n0 = blockIdx.x * 32;
#pragma unroll
  for (int i = 0; i < 4; ++i)
    T[ty + 8 * i][tx] = f2bf(W[(size_t)(k0 + ty + 8 * i) * N + n0 + tx]);
  __syncthreads();
#pragma unroll
  for (int i = 0; i < 4; ++i)
    Wt[(size_t)(n0 + ty + 8 * i) * K + k0 + tx] = T[tx][ty + 8 * i];
}

// ---------------------------------------------------------------------------
// LayerNorm (D=1024) fused with bf16 cast. One block per row, 256 threads.
// ---------------------------------------------------------------------------
__global__ __launch_bounds__(256) void ln_kernel(const float* __restrict__ x,
                                                 const float* __restrict__ g,
                                                 const float* __restrict__ b,
                                                 u16* __restrict__ y) {
  const int row = blockIdx.x;
  const int tid = threadIdx.x;
  const float* xr = x + (size_t)row * 1024;
  float v[4];
  float s = 0.f, s2 = 0.f;
#pragma unroll
  for (int i = 0; i < 4; ++i) {
    v[i] = xr[tid + i * 256];
    s += v[i];
    s2 += v[i] * v[i];
  }
#pragma unroll
  for (int m = 16; m >= 1; m >>= 1) {
    s  += __shfl_xor(s, m, 32);
    s2 += __shfl_xor(s2, m, 32);
  }
  __shared__ float rs[8], rq[8];
  if ((tid & 31) == 0) { rs[tid >> 5] = s; rq[tid >> 5] = s2; }
  __syncthreads();
  float ts = 0.f, tq = 0.f;
#pragma unroll
  for (int w = 0; w < 8; ++w) { ts += rs[w]; tq += rq[w]; }
  const float mu   = ts * (1.0f / 1024.0f);
  const float var  = tq * (1.0f / 1024.0f) - mu * mu;
  const float rstd = rsqrtf(var + 1e-5f);
#pragma unroll
  for (int i = 0; i < 4; ++i) {
    const int c = tid + i * 256;
    y[(size_t)row * 1024 + c] = f2bf((v[i] - mu) * rstd * g[c] + b[c]);
  }
}

// ---------------------------------------------------------------------------
// Tiled WMMA GEMM: C[M,N] = A_bf16[M,K] @ Wt_bf16[N,K]^T + bias (+epilogue)
// BM=128 BN=128 BK=32, 8 waves, each wave: 32x64 tile (2x4 wmma tiles).
// Tiles staged by TDM (double-buffered) when available.
// EPI: 0 = bias -> bf16 out, 1 = bias+gelu_new -> bf16 out,
//      2 = bias+residual -> f32 out
// ---------------------------------------------------------------------------
template <int EPI>
__global__ __launch_bounds__(256) void gemm_kernel(
    const u16* __restrict__ A, const u16* __restrict__ Bt,
    const float* __restrict__ bias, const float* __restrict__ res,
    u16* __restrict__ outB, float* __restrict__ outF,
    int M, int N, int K) {
#if HAVE_TDM
  __shared__ u16 As[2][128][40];   // [m][k], +8 halfs pad = TDM pad_amount
  __shared__ u16 Bs[2][128][40];   // [n][k]
#else
  __shared__ u16 As[1][128][40];
  __shared__ u16 Bs[1][128][40];
#endif

  const int tid  = threadIdx.x;
  const int lane = tid & 31;
  const int wave = tid >> 5;
  const int wm   = wave >> 1;   // 0..3
  const int wn   = wave & 1;    // 0..1
  const int half = lane >> 4;
  const int l15  = lane & 15;
  const int bm = blockIdx.y * 128;
  const int bn = blockIdx.x * 128;

  const v8f zero = {0.f, 0.f, 0.f, 0.f, 0.f, 0.f, 0.f, 0.f};
  v8f acc[2][4];
#pragma unroll
  for (int i = 0; i < 2; ++i)
#pragma unroll
    for (int j = 0; j < 4; ++j) acc[i][j] = zero;

  const int nk = K >> 5;

#if HAVE_TDM
  if (wave == 0) {
    tdm_load_tile_128x32((u32)(uintptr_t)&As[0][0][0], A + (size_t)bm * K, K, M);
    tdm_load_tile_128x32((u32)(uintptr_t)&Bs[0][0][0], Bt + (size_t)bn * K, K, N);
  }
#else
  const int arow = tid >> 1;          // 0..127
  const int aseg = (tid & 1) * 16;    // 0 or 16
#endif

  for (int it = 0; it < nk; ++it) {
#if HAVE_TDM
    const int cur = it & 1;
    if (wave == 0) __builtin_amdgcn_s_wait_tensorcnt(0);
    __syncthreads();                   // tile `it` visible to all waves
    if (wave == 0 && it + 1 < nk) {    // prefetch next tile into other buffer
      const int k0 = (it + 1) << 5;
      tdm_load_tile_128x32((u32)(uintptr_t)&As[cur ^ 1][0][0],
                           A + (size_t)bm * K + k0, K, M);
      tdm_load_tile_128x32((u32)(uintptr_t)&Bs[cur ^ 1][0][0],
                           Bt + (size_t)bn * K + k0, K, N);
    }
#else
    const int cur = 0;
    const int k0 = it << 5;
    __syncthreads();
    {
      const u16* ga = A + (size_t)(bm + arow) * K + k0 + aseg;
      *(uint4*)&As[0][arow][aseg]     = *(const uint4*)ga;
      *(uint4*)&As[0][arow][aseg + 8] = *(const uint4*)(ga + 8);
      const u16* gb = Bt + (size_t)(bn + arow) * K + k0 + aseg;
      *(uint4*)&Bs[0][arow][aseg]     = *(const uint4*)gb;
      *(uint4*)&Bs[0][arow][aseg + 8] = *(const uint4*)(gb + 8);
    }
    __syncthreads();
#endif

    Frag af[2], bf[4];
#pragma unroll
    for (int mt = 0; mt < 2; ++mt) {
      // 16-bit A layout: lane<16 -> K {0..7,16..23}; lane>=16 -> K {8..15,24..31}
      const int r = wm * 32 + mt * 16 + l15;
      af[mt].q[0] = *(const uint4*)&As[cur][r][half * 8];
      af[mt].q[1] = *(const uint4*)&As[cur][r][16 + half * 8];
    }
#pragma unroll
    for (int nt = 0; nt < 4; ++nt) {
      // B layout: lane = column N, K {0..7,16..23} (lane<16) / {8..15,24..31} (lane>=16)
      const int n = wn * 64 + nt * 16 + l15;
      bf[nt].q[0] = *(const uint4*)&Bs[cur][n][half * 8];
      bf[nt].q[1] = *(const uint4*)&Bs[cur][n][16 + half * 8];
    }
#pragma unroll
    for (int mt = 0; mt < 2; ++mt)
#pragma unroll
      for (int nt = 0; nt < 4; ++nt)
        acc[mt][nt] = wmma_bf16(af[mt], bf[nt], acc[mt][nt]);
  }

  // Epilogue: C fragment layout -> row = r + 8*half, col = l15 within tile
#pragma unroll
  for (int mt = 0; mt < 2; ++mt)
#pragma unroll
    for (int nt = 0; nt < 4; ++nt)
#pragma unroll
      for (int r = 0; r < 8; ++r) {
        const int row = bm + wm * 32 + mt * 16 + r + half * 8;
        const int col = bn + wn * 64 + nt * 16 + l15;
        float v = acc[mt][nt][r] + bias[col];
        const size_t idx = (size_t)row * N + col;
        if (EPI == 0) {
          outB[idx] = f2bf(v);
        } else if (EPI == 1) {
          const float t = v + 0.044715f * v * v * v;
          const float th = 2.0f / (1.0f + __expf(-1.5957691216f * t)) - 1.0f;
          outB[idx] = f2bf(0.5f * v * (1.0f + th));
        } else {
          outF[idx] = v + res[idx];
        }
      }
}

// ---------------------------------------------------------------------------
// Flash attention: qkv_bf16 [B,S,3D] -> ctx_bf16 [B,S,D] (heads merged)
// grid = (S/128, B*H); 8 waves, each wave owns 16 query rows.
// ---------------------------------------------------------------------------
__global__ __launch_bounds__(256) void attn_kernel(const u16* __restrict__ qkv,
                                                   u16* __restrict__ ctx) {
  __shared__ u16 Ks[64][72];        // [key][dh], row-major
  __shared__ u16 Vt[64][72];        // [dh][key], transposed
  __shared__ u16 Ps[8][16][72];     // per-wave P relayout strip

  const int tid  = threadIdx.x;
  const int lane = tid & 31, wave = tid >> 5;
  const int half = lane >> 4, l15 = lane & 15;
  const int bh = blockIdx.y;
  const int bb = bh >> 4, hh = bh & 15;
  const int q0 = blockIdx.x * 128 + wave * 16;

  const u16* base = qkv + (size_t)bb * 2048 * 3072 + hh * 64;

  Frag aq[2];
  {
    const u16* qp = base + (size_t)(q0 + l15) * 3072;
#pragma unroll
    for (int f = 0; f < 2; ++f) {
      aq[f].q[0] = *(const uint4*)(qp + f * 32 + half * 8);
      aq[f].q[1] = *(const uint4*)(qp + f * 32 + 16 + half * 8);
    }
  }

  const v8f zero = {0.f, 0.f, 0.f, 0.f, 0.f, 0.f, 0.f, 0.f};
  v8f o[4];
#pragma unroll
  for (int nt = 0; nt < 4; ++nt) o[nt] = zero;
  float mi[8], li[8];
#pragma unroll
  for (int r = 0; r < 8; ++r) { mi[r] = -3.0e38f; li[r] = 0.f; }

  const int kkey = tid >> 2;            // 0..63
  const int kdh  = (tid & 3) * 16;      // 0..48
  const int nkb  = blockIdx.x * 2 + 2;  // 64-key blocks covering causal span

  for (int j = 0; j < nkb; ++j) {
    __syncthreads();
    {  // cooperative stage of K (row-major) and V (transposed)
      const u16* kp = base + 1024 + (size_t)(j * 64 + kkey) * 3072 + kdh;
      *(uint4*)&Ks[kkey][kdh]     = *(const uint4*)kp;
      *(uint4*)&Ks[kkey][kdh + 8] = *(const uint4*)(kp + 8);
      const u16* vp = base + 2048 + (size_t)(j * 64 + kkey) * 3072 + kdh;
      uint4 t0 = *(const uint4*)vp;
      uint4 t1 = *(const uint4*)(vp + 8);
      u16 tmp[16];
      *(uint4*)&tmp[0] = t0;
      *(uint4*)&tmp[8] = t1;
#pragma unroll
      for (int i = 0; i < 16; ++i) Vt[kdh + i][kkey] = tmp[i];
    }
    __syncthreads();
    if (j * 64 > q0 + 15) continue;  // wave-uniform; barriers stay matched

    v8f sc[4];
#pragma unroll
    for (int nt = 0; nt < 4; ++nt) {
      sc[nt] = zero;
      const int n = nt * 16 + l15;
#pragma unroll
      for (int f = 0; f < 2; ++f) {
        Frag bk;
        bk.q[0] = *(const uint4*)&Ks[n][f * 32 + half * 8];
        bk.q[1] = *(const uint4*)&Ks[n][f * 32 + 16 + half * 8];
        sc[nt] = wmma_bf16(aq[f], bk, sc[nt]);
      }
    }

#pragma unroll
    for (int r = 0; r < 8; ++r) {
      const int qrow = q0 + r + half * 8;
      float mloc = -3.0e38f;
#pragma unroll
      for (int nt = 0; nt < 4; ++nt) {
        const int col = j * 64 + nt * 16 + l15;
        float s = sc[nt][r] * 0.125f;          // 1/sqrt(64)
        if (col > qrow) s = -3.0e38f;          // causal mask
        sc[nt][r] = s;
        mloc = fmaxf(mloc, s);
      }
#pragma unroll
      for (int m = 8; m >= 1; m >>= 1) mloc = fmaxf(mloc, __shfl_xor(mloc, m, 32));
      const float mnew = fmaxf(mi[r], mloc);
      const float co = __expf(mi[r] - mnew);
      mi[r] = mnew;
      float rsum = 0.f;
#pragma unroll
      for (int nt = 0; nt < 4; ++nt) {
        const float p = __expf(sc[nt][r] - mnew);
        sc[nt][r] = p;
        rsum += p;
      }
#pragma unroll
      for (int m = 8; m >= 1; m >>= 1) rsum += __shfl_xor(rsum, m, 32);
      li[r] = li[r] * co + rsum;
#pragma unroll
      for (int nt = 0; nt < 4; ++nt) o[nt][r] = o[nt][r] * co;
    }

    // relayout P: C-fragment -> A-fragment via per-wave LDS strip
#pragma unroll
    for (int nt = 0; nt < 4; ++nt)
#pragma unroll
      for (int r = 0; r < 8; ++r)
        Ps[wave][r + half * 8][nt * 16 + l15] = f2bf(sc[nt][r]);

    asm volatile("s_wait_dscnt 0x0" ::: "memory");  // lockstep wave relayout

    Frag ap[2];
#pragma unroll
    for (int f = 0; f < 2; ++f) {
      ap[f].q[0] = *(const uint4*)&Ps[wave][l15][f * 32 + half * 8];
      ap[f].q[1] = *(const uint4*)&Ps[wave][l15][f * 32 + 16 + half * 8];
    }

#pragma unroll
    for (int nt = 0; nt < 4; ++nt) {
      const int n = nt * 16 + l15;
#pragma unroll
      for (int f = 0; f < 2; ++f) {
        Frag bv;
        bv.q[0] = *(const uint4*)&Vt[n][f * 32 + half * 8];
        bv.q[1] = *(const uint4*)&Vt[n][f * 32 + 16 + half * 8];
        o[nt] = wmma_bf16(ap[f], bv, o[nt]);
      }
    }
  }

#pragma unroll
  for (int r = 0; r < 8; ++r) {
    const float inv = 1.0f / li[r];
    const int row = q0 + r + half * 8;
#pragma unroll
    for (int nt = 0; nt < 4; ++nt) {
      const int col = hh * 64 + nt * 16 + l15;
      ctx[((size_t)bb * 2048 + row) * 1024 + col] = f2bf(o[nt][r] * inv);
    }
  }
}

// ---------------------------------------------------------------------------
// Launch
// ---------------------------------------------------------------------------
extern "C" void kernel_launch(void* const* d_in, const int* in_sizes, int n_in,
                              void* d_out, int out_size, void* d_ws, size_t ws_size,
                              hipStream_t stream) {
  (void)in_sizes; (void)n_in; (void)out_size; (void)ws_size;
  const float* x        = (const float*)d_in[0];
  const float* ln1_g    = (const float*)d_in[1];
  const float* ln1_b    = (const float*)d_in[2];
  const float* c_attn_w = (const float*)d_in[3];
  const float* c_attn_b = (const float*)d_in[4];
  const float* c_proj_w = (const float*)d_in[5];
  const float* c_proj_b = (const float*)d_in[6];
  const float* ln2_g    = (const float*)d_in[7];
  const float* ln2_b    = (const float*)d_in[8];
  const float* fc_w     = (const float*)d_in[9];
  const float* fc_b     = (const float*)d_in[10];
  const float* proj_w   = (const float*)d_in[11];
  const float* proj_b   = (const float*)d_in[12];
  float* out = (float*)d_out;

  char* p = (char*)d_ws;
  u16* wqkv = (u16*)p; p += (size_t)1024 * 3072 * 2;   // c_attn_w^T bf16 [3072][1024]
  u16* wcp  = (u16*)p; p += (size_t)1024 * 1024 * 2;   // c_proj_w^T bf16 [1024][1024]
  u16* wfc  = (u16*)p; p += (size_t)1024 * 4096 * 2;   // fc_w^T bf16 [4096][1024]
  u16* wp2  = (u16*)p; p += (size_t)4096 * 1024 * 2;   // proj_w^T bf16 [1024][4096]
  u16* h1   = (u16*)p; p += (size_t)4096 * 1024 * 2;   // ln1 out
  u16* qkva = (u16*)p; p += (size_t)4096 * 3072 * 2;   // qkv activations
  u16* ctx  = (u16*)p; p += (size_t)4096 * 1024 * 2;   // attention context
  float* x2 = (float*)p; p += (size_t)4096 * 1024 * 4; // after attn residual
  u16* h2   = (u16*)p; p += (size_t)4096 * 1024 * 2;   // ln2 out
  u16* a1   = (u16*)p; p += (size_t)4096 * 4096 * 2;   // gelu(fc) out

  // weight cast + transpose: W[K][N] -> Wt[N][K]
  transpose_cast_kernel<<<dim3(96, 32),  256, 0, stream>>>(c_attn_w, wqkv, 1024, 3072);
  transpose_cast_kernel<<<dim3(32, 32),  256, 0, stream>>>(c_proj_w, wcp,  1024, 1024);
  transpose_cast_kernel<<<dim3(128, 32), 256, 0, stream>>>(fc_w,     wfc,  1024, 4096);
  transpose_cast_kernel<<<dim3(32, 128), 256, 0, stream>>>(proj_w,   wp2,  4096, 1024);

  ln_kernel<<<4096, 256, 0, stream>>>(x, ln1_g, ln1_b, h1);
  gemm_kernel<0><<<dim3(24, 32), 256, 0, stream>>>(h1, wqkv, c_attn_b, nullptr,
                                                   qkva, nullptr, 4096, 3072, 1024);
  attn_kernel<<<dim3(16, 32), 256, 0, stream>>>(qkva, ctx);
  gemm_kernel<2><<<dim3(8, 32), 256, 0, stream>>>(ctx, wcp, c_proj_b, x,
                                                  nullptr, x2, 4096, 1024, 1024);
  ln_kernel<<<4096, 256, 0, stream>>>(x2, ln2_g, ln2_b, h2);
  gemm_kernel<1><<<dim3(32, 32), 256, 0, stream>>>(h2, wfc, fc_b, nullptr,
                                                   a1, nullptr, 4096, 4096, 1024);
  gemm_kernel<2><<<dim3(8, 32), 256, 0, stream>>>(a1, wp2, proj_b, x2,
                                                  nullptr, out, 4096, 1024, 4096);
}
